// plgMultiHeadAttention_28217935135476
// MI455X (gfx1250) — compile-verified, matrix operand order checked by reading, not run on hardware
//
#include <hip/hip_runtime.h>
#include <hip/hip_bf16.h>

// ---------------------------------------------------------------------------
// Problem constants (from reference)
// ---------------------------------------------------------------------------
constexpr int CB  = 2;       // batch
constexpr int CS  = 2048;    // sequence
constexpr int CDM = 1024;    // model dim
constexpr int CH  = 16;      // heads
constexpr int CD  = 64;      // head dim
constexpr int CAD = 1024;    // A_DEPTH
constexpr int CNRES = 8;
constexpr int CNGLU = 2;

typedef __attribute__((ext_vector_type(16))) __bf16 v16bf;
typedef __attribute__((ext_vector_type(8)))  float  v8f;
typedef int i32x4_ __attribute__((ext_vector_type(4)));
typedef i32x4_ __attribute__((address_space(1))) as1_i32x4;
typedef i32x4_ __attribute__((address_space(3))) as3_i32x4;

// --- CDNA5 async global->LDS path (ASYNCcnt), with safe fallback -----------
#if defined(__has_builtin)
#  if __has_builtin(__builtin_amdgcn_global_load_async_to_lds_b128) && \
      __has_builtin(__builtin_amdgcn_s_wait_asynccnt)
#    define HAS_ASYNC_LDS 1
#  endif
#endif
#ifndef HAS_ASYNC_LDS
#  define HAS_ASYNC_LDS 0
#endif

__device__ __forceinline__ void async_cp16(void* lds_dst, const void* g_src) {
#if HAS_ASYNC_LDS
  __builtin_amdgcn_global_load_async_to_lds_b128(
      (as1_i32x4*)(g_src), (as3_i32x4*)(lds_dst), 0, 0);
#else
  *(uint4*)lds_dst = *(const uint4*)g_src;
#endif
}
__device__ __forceinline__ void async_wait0() {
#if HAS_ASYNC_LDS
  __builtin_amdgcn_s_wait_asynccnt(0);
#endif
}

union FragBF { v16bf v; uint4 q[2]; };

__device__ __forceinline__ v8f wmma_bf16(const FragBF& a, const FragBF& b, v8f c) {
  return __builtin_amdgcn_wmma_f32_16x16x32_bf16(false, a.v, false, b.v,
                                                 (short)0, c, false, false);
}

// Load a 16x32 A-fragment / B^T-fragment from LDS (row-major, leading dim ld):
// lane (l&15) picks the row, (l>>4) picks the K half-group (per ISA 7.12.2).
__device__ __forceinline__ FragBF load_frag(const __bf16* base, int ld, int row,
                                            int kk, int g) {
  FragBF f;
  const __bf16* p = base + row * ld + kk + g * 8;
  f.q[0] = *(const uint4*)(p);
  f.q[1] = *(const uint4*)(p + 16);
  return f;
}

// ---------------------------------------------------------------------------
// Generic batched bf16 WMMA GEMM (64x64 tile) for small/batched shapes.
// A: bf16 row-major (lda == K).  BT: B transposed = [N,K] bf16 (ld == K).
// M,N multiples of 64; K multiple of 32.  8 waves, 2 wmma tiles/wave.
// ---------------------------------------------------------------------------
__global__ __launch_bounds__(256)
void k_gemm_bf16(const __bf16* __restrict__ A, const __bf16* __restrict__ BT,
                 const float* __restrict__ bias,
                 float* __restrict__ C, __bf16* __restrict__ Cbf,
                 int M, int N, int K,
                 long long sA, long long sBT, long long sC)
{
  const int batch = blockIdx.z;
  A  += (long long)batch * sA;
  BT += (long long)batch * sBT;
  const long long cbase = (long long)batch * sC;

  const int m0 = blockIdx.y * 64;
  const int n0 = blockIdx.x * 64;

  __shared__ __align__(16) __bf16 As[2][64 * 32];
  __shared__ __align__(16) __bf16 Bs[2][64 * 32];

  const int tid  = threadIdx.x;
  const int wave = tid >> 5, lane = tid & 31;
  const int wm = wave >> 2, wn = wave & 3;   // 2 x 4 wave grid
  const int g  = lane >> 4, lr = lane & 15;

  v8f acc0 = {};
  v8f acc1 = {};

  const int r  = tid >> 2;            // 0..63 tile row
  const int c8 = (tid & 3) << 3;      // 0,8,16,24 (bf16 elements)
  const __bf16* gA = A  + (long long)(m0 + r) * K + c8;
  const __bf16* gB = BT + (long long)(n0 + r) * K + c8;
  const int lidx = r * 32 + c8;

  // prologue: stage tile 0 into buffer 0
  async_cp16(&As[0][lidx], gA);
  async_cp16(&Bs[0][lidx], gB);

  int buf = 0;
  for (int k0 = 0; k0 < K; k0 += 32) {
    async_wait0();          // my DMA into `buf` has landed
    __syncthreads();        // everyone's has; everyone done reading buf^1

    if (k0 + 32 < K) {      // kick off next tile into the other buffer
      async_cp16(&As[buf ^ 1][lidx], gA + k0 + 32);
      async_cp16(&Bs[buf ^ 1][lidx], gB + k0 + 32);
#if !HAS_ASYNC_LDS
      __builtin_prefetch(gA + k0 + 64, 0, 3);
      __builtin_prefetch(gB + k0 + 64, 0, 3);
#endif
    }

    const int mA = wm * 16 + lr;
    const int nB = wn * 16 + lr;
    FragBF a0 = load_frag(As[buf], 32, mA,      0, g);
    FragBF a1 = load_frag(As[buf], 32, mA + 32, 0, g);
    FragBF b  = load_frag(Bs[buf], 32, nB,      0, g);
    acc0 = wmma_bf16(a0, b, acc0);
    acc1 = wmma_bf16(a1, b, acc1);
    buf ^= 1;
  }

  const int col = n0 + wn * 16 + lr;
  const float bv = bias ? bias[col] : 0.f;
#pragma unroll
  for (int v = 0; v < 8; ++v) {
    const int row0 = m0 + wm * 16 + g * 8 + v;   // C/D layout: M = v + 8*(lane>=16)
    const float o0 = acc0[v] + bv;
    const float o1 = acc1[v] + bv;
    const long long i0 = cbase + (long long)row0 * N + col;
    const long long i1 = cbase + (long long)(row0 + 32) * N + col;
    if (C)   { C[i0] = o0;            C[i1] = o1; }
    if (Cbf) { Cbf[i0] = (__bf16)o0;  Cbf[i1] = (__bf16)o1; }
  }
}

// ---------------------------------------------------------------------------
// Big-tile bf16 WMMA GEMM (128x128 tile) for the dominant GEMMs.
// Each of the 8 waves (2x4 grid) owns a 64x32 C block = 4x2 WMMA tiles,
// so 8 v_wmma per k-step from 6 fragment loads; LDS staged by async DMA,
// double buffered (32 KB).  M,N multiples of 128; K multiple of 32.
// ---------------------------------------------------------------------------
__global__ __launch_bounds__(256)
void k_gemm_bf16_128(const __bf16* __restrict__ A, const __bf16* __restrict__ BT,
                     const float* __restrict__ bias,
                     float* __restrict__ C, __bf16* __restrict__ Cbf,
                     int M, int N, int K,
                     long long sA, long long sBT, long long sC)
{
  const int batch = blockIdx.z;
  A  += (long long)batch * sA;
  BT += (long long)batch * sBT;
  const long long cbase = (long long)batch * sC;

  const int m0 = blockIdx.y * 128;
  const int n0 = blockIdx.x * 128;

  __shared__ __align__(16) __bf16 As[2][128 * 32];
  __shared__ __align__(16) __bf16 Bs[2][128 * 32];

  const int tid  = threadIdx.x;
  const int wave = tid >> 5, lane = tid & 31;
  const int wm = wave >> 2, wn = wave & 3;   // wave covers rows wm*64.., cols wn*32..
  const int g  = lane >> 4, lr = lane & 15;

  v8f acc[4][2];
#pragma unroll
  for (int i = 0; i < 4; ++i)
#pragma unroll
    for (int j = 0; j < 2; ++j) acc[i][j] = {};

  const int r   = tid >> 1;            // 0..127 tile row
  const int c16 = (tid & 1) << 4;      // 0 or 16 (bf16 elements)
  const __bf16* gA = A  + (long long)(m0 + r) * K + c16;
  const __bf16* gB = BT + (long long)(n0 + r) * K + c16;
  const int lidx = r * 32 + c16;

  // prologue: stage tile 0
  async_cp16(&As[0][lidx],     gA);
  async_cp16(&As[0][lidx + 8], gA + 8);
  async_cp16(&Bs[0][lidx],     gB);
  async_cp16(&Bs[0][lidx + 8], gB + 8);

  int buf = 0;
  for (int k0 = 0; k0 < K; k0 += 32) {
    async_wait0();
    __syncthreads();

    if (k0 + 32 < K) {
      const __bf16* nA = gA + k0 + 32;
      const __bf16* nB = gB + k0 + 32;
      async_cp16(&As[buf ^ 1][lidx],     nA);
      async_cp16(&As[buf ^ 1][lidx + 8], nA + 8);
      async_cp16(&Bs[buf ^ 1][lidx],     nB);
      async_cp16(&Bs[buf ^ 1][lidx + 8], nB + 8);
#if !HAS_ASYNC_LDS
      __builtin_prefetch(gA + k0 + 64, 0, 3);
      __builtin_prefetch(gB + k0 + 64, 0, 3);
#endif
    }

    FragBF a[4], b[2];
#pragma unroll
    for (int i = 0; i < 4; ++i)
      a[i] = load_frag(As[buf], 32, wm * 64 + i * 16 + lr, 0, g);
#pragma unroll
    for (int j = 0; j < 2; ++j)
      b[j] = load_frag(Bs[buf], 32, wn * 32 + j * 16 + lr, 0, g);
#pragma unroll
    for (int j = 0; j < 2; ++j)
#pragma unroll
      for (int i = 0; i < 4; ++i)
        acc[i][j] = wmma_bf16(a[i], b[j], acc[i][j]);
    buf ^= 1;
  }

#pragma unroll
  for (int j = 0; j < 2; ++j) {
    const int col = n0 + wn * 32 + j * 16 + lr;
    const float bv = bias ? bias[col] : 0.f;
#pragma unroll
    for (int i = 0; i < 4; ++i) {
#pragma unroll
      for (int v = 0; v < 8; ++v) {
        const int row = m0 + wm * 64 + i * 16 + g * 8 + v;
        const float o = acc[i][j][v] + bv;
        const long long ix = cbase + (long long)row * N + col;
        if (C)   C[ix] = o;
        if (Cbf) Cbf[ix] = (__bf16)o;
      }
    }
  }
}

// ---------------------------------------------------------------------------
// Flash attention with causal masking and online softmax.
// Q (=qA) [BH,S,D] bf16, Kh [BH,S,D] bf16, VT [BH,D,S] bf16 -> O [B,S,DM] bf16
// Block: 64 query rows, 8 waves; j-tiles of 64 keys; never materializes SxS.
// K/V tiles staged via the async global->LDS engine.
// ---------------------------------------------------------------------------
__global__ __launch_bounds__(256)
void k_flash(const __bf16* __restrict__ Q, const __bf16* __restrict__ Kh,
             const __bf16* __restrict__ VT, const float* __restrict__ ba,
             __bf16* __restrict__ O)
{
  const int i0 = blockIdx.x * 64;
  const int bh = blockIdx.y;
  const int b  = bh >> 4, h = bh & 15;

  __shared__ __align__(16) __bf16 Qs [64 * 64];
  __shared__ __align__(16) __bf16 Ks [64 * 64];
  __shared__ __align__(16) __bf16 VTs[64 * 64];
  __shared__ __align__(16) float  Ss [64 * 64];
  __shared__ __align__(16) __bf16 Ps [64 * 64];
  __shared__ float mrow[64], lrow[64], arow[64], red[64 * 4];

  const int tid  = threadIdx.x;
  const int wave = tid >> 5, lane = tid & 31;
  const int wm = wave >> 2, wn = wave & 3;
  const int g  = lane >> 4, lr = lane & 15;
  const float bav   = ba[0];
  const float scale = 0.125f;   // 1/sqrt(D=64)

  const int e  = tid * 16;      // 16 bf16 elements staged per thread
  const int rr = e >> 6, cc = e & 63;

  { // stage Q tile (8 KB) via async DMA
    const __bf16* src = Q + ((long long)bh * CS + (i0 + rr)) * CD + cc;
    async_cp16(&Qs[e],     src);
    async_cp16(&Qs[e + 8], src + 8);
  }
  if (tid < 64) { mrow[tid] = -INFINITY; lrow[tid] = 0.f; }

  v8f o0 = {};
  v8f o1 = {};
  async_wait0();
  __syncthreads();

  for (int j0 = 0; j0 <= i0; j0 += 64) {   // causal: skip j-tiles above diagonal
    { // stage K tile (row-major = B^T layout for QK^T) and V^T tile
      const __bf16* ks = Kh + ((long long)bh * CS + (j0 + rr)) * CD + cc;
      async_cp16(&Ks[e],      ks);
      async_cp16(&Ks[e + 8],  ks + 8);
      const __bf16* vs = VT + ((long long)bh * CD + rr) * CS + j0 + cc;
      async_cp16(&VTs[e],     vs);
      async_cp16(&VTs[e + 8], vs + 8);
    }
    async_wait0();
    __syncthreads();

    // S = Q K^T over head dim (K=64 -> two k-chunks of 32)
    v8f s0 = {};
    v8f s1 = {};
    const int mA = wm * 16 + lr;
    const int nB = wn * 16 + lr;
#pragma unroll
    for (int kk = 0; kk < 64; kk += 32) {
      FragBF a0 = load_frag(Qs, 64, mA,      kk, g);
      FragBF a1 = load_frag(Qs, 64, mA + 32, kk, g);
      FragBF bf = load_frag(Ks, 64, nB,      kk, g);
      s0 = wmma_bf16(a0, bf, s0);
      s1 = wmma_bf16(a1, bf, s1);
    }
    { // scale + bias + causal mask, spill to LDS for row-wise softmax
      const int col = wn * 16 + lr;
#pragma unroll
      for (int v = 0; v < 8; ++v) {
        const int r0 = wm * 16 + g * 8 + v;
        float e0 = s0[v] * scale + bav;
        float e1 = s1[v] * scale + bav;
        if (j0 + col > i0 + r0)      e0 = -1e9f;
        if (j0 + col > i0 + r0 + 32) e1 = -1e9f;
        Ss[r0 * 64 + col]        = e0;
        Ss[(r0 + 32) * 64 + col] = e1;
      }
    }
    __syncthreads();

    { // partial row max (4 threads / row)
      const int rw = tid >> 2, c0 = (tid & 3) << 4;
      float lm = -INFINITY;
#pragma unroll
      for (int c = 0; c < 16; ++c) lm = fmaxf(lm, Ss[rw * 64 + c0 + c]);
      red[rw * 4 + (tid & 3)] = lm;
    }
    __syncthreads();
    if (tid < 64) { // online-softmax stats update
      const float tm = fmaxf(fmaxf(red[tid * 4 + 0], red[tid * 4 + 1]),
                             fmaxf(red[tid * 4 + 2], red[tid * 4 + 3]));
      const float mo = mrow[tid];
      const float mn = fmaxf(mo, tm);
      arow[tid] = __expf(mo - mn);      // exp(-inf)=0 on first tile
      mrow[tid] = mn;
    }
    __syncthreads();

    { // P = exp(S-m) in bf16 + partial sums
      const int rw = tid >> 2, c0 = (tid & 3) << 4;
      const float mn = mrow[rw];
      float ps = 0.f;
#pragma unroll
      for (int c = 0; c < 16; ++c) {
        const float p = __expf(Ss[rw * 64 + c0 + c] - mn);
        Ps[rw * 64 + c0 + c] = (__bf16)p;
        ps += p;
      }
      red[rw * 4 + (tid & 3)] = ps;
    }
    { // rescale running O accumulators
#pragma unroll
      for (int v = 0; v < 8; ++v) {
        const int r0 = wm * 16 + g * 8 + v;
        o0[v] *= arow[r0];
        o1[v] *= arow[r0 + 32];
      }
    }
    __syncthreads();
    if (tid < 64)
      lrow[tid] = arow[tid] * lrow[tid] +
                  (red[tid * 4 + 0] + red[tid * 4 + 1] +
                   red[tid * 4 + 2] + red[tid * 4 + 3]);

    // O += P V  (V^T tile already in B-fragment layout [d][j])
#pragma unroll
    for (int kk = 0; kk < 64; kk += 32) {
      FragBF a0 = load_frag(Ps,  64, mA,      kk, g);
      FragBF a1 = load_frag(Ps,  64, mA + 32, kk, g);
      FragBF bf = load_frag(VTs, 64, nB,      kk, g);
      o0 = wmma_bf16(a0, bf, o0);
      o1 = wmma_bf16(a1, bf, o1);
    }
    __syncthreads();
  }

  { // normalize, emit [B,S,DM] bf16 for the output projection GEMM
    const int col = wn * 16 + lr;
#pragma unroll
    for (int v = 0; v < 8; ++v) {
      const int r0 = wm * 16 + g * 8 + v;
      const float n0v = o0[v] / lrow[r0];
      const float n1v = o1[v] / lrow[r0 + 32];
      const long long a0i = (((long long)b * CS) + i0 + r0)      * CDM + h * CD + col;
      const long long a1i = (((long long)b * CS) + i0 + r0 + 32) * CDM + h * CD + col;
      O[a0i] = (__bf16)n0v;
      O[a1i] = (__bf16)n1v;
    }
  }
}

// ---------------------------------------------------------------------------
// Elementwise / prep kernels
// ---------------------------------------------------------------------------
__global__ __launch_bounds__(256)
void k_convT(const float* __restrict__ W, __bf16* __restrict__ WT, int Kd, int Nd) {
  const long long idx = (long long)blockIdx.x * 256 + threadIdx.x;
  if (idx >= (long long)Kd * Nd) return;
  const int k = (int)(idx / Nd), n = (int)(idx % Nd);
  WT[(long long)n * Kd + k] = (__bf16)W[idx];
}

__global__ __launch_bounds__(256)
void k_conv(const float* __restrict__ x, __bf16* __restrict__ y, long long n) {
  const long long idx = (long long)blockIdx.x * 256 + threadIdx.x;
  if (idx < n) y[idx] = (__bf16)x[idx];
}

// RoPE + head split.  Emits qh [BH,S,D], qhT [BH,D,S], kh [BH,S,D], vT [BH,D,S].
__global__ __launch_bounds__(256)
void k_qkv_prep(const float* __restrict__ qp, const float* __restrict__ kp,
                const float* __restrict__ vp,
                __bf16* __restrict__ qh, __bf16* __restrict__ qhT,
                __bf16* __restrict__ kh, __bf16* __restrict__ vT)
{
  const long long idx = (long long)blockIdx.x * 256 + threadIdx.x; // B*S*DM
  const int d = (int)(idx & 63);
  const int h = (int)((idx >> 6) & 15);
  const int s = (int)((idx >> 10) & 2047);
  const int b = (int)(idx >> 21);
  const int bh = b * CH + h;
  const int dd = (d < 32) ? d : d - 32;
  const float invf = __powf(10000.f, -(float)dd * (1.f / 32.f));
  float cs, sn;
  __sincosf((float)s * invf, &sn, &cs);   // (sin, cos)
  const long long base = ((long long)(b * CS + s)) * CDM + h * CD;
  const float qx = qp[base + d];
  const float qo = (d < 32) ? qp[base + d + 32] : qp[base + d - 32];
  const float qr = (d < 32) ? (qx * cs - qo * sn) : (qx * cs + qo * sn);
  const float kx = kp[base + d];
  const float ko = (d < 32) ? kp[base + d + 32] : kp[base + d - 32];
  const float kr = (d < 32) ? (kx * cs - ko * sn) : (kx * cs + ko * sn);
  const float vv = vp[base + d];
  const long long rm = ((long long)bh * CS + s) * CD + d;   // [bh][s][d]
  const long long tm = ((long long)bh * CD + d) * CS + s;   // [bh][d][s]
  qh[rm]  = (__bf16)qr;
  qhT[tm] = (__bf16)qr;
  kh[rm]  = (__bf16)kr;
  vT[tm]  = (__bf16)vv;
}

// LayerNorm over rows of length C (<=1024), optional residual, f32 + bf16 out.
__global__ __launch_bounds__(256)
void k_ln(const float* __restrict__ x, const float* __restrict__ res,
          const float* __restrict__ gamma, const float* __restrict__ beta,
          float* __restrict__ y, __bf16* __restrict__ ybf, int C)
{
  const long long row = blockIdx.x;
  const float* xr = x + row * C;
  const float* rr = res ? res + row * C : nullptr;
  float vloc[4];
  int cnt = 0;
  float s = 0.f;
  for (int c = threadIdx.x; c < C; c += 256) {
    float v = xr[c] + (rr ? rr[c] : 0.f);
    vloc[cnt++] = v;
    s += v;
  }
  __shared__ float sh[256];
  sh[threadIdx.x] = s;
  __syncthreads();
  for (int o = 128; o > 0; o >>= 1) {
    if (threadIdx.x < o) sh[threadIdx.x] += sh[threadIdx.x + o];
    __syncthreads();
  }
  const float mean = sh[0] / C;
  __syncthreads();
  float s2 = 0.f;
  for (int i = 0; i < cnt; ++i) { const float d = vloc[i] - mean; s2 += d * d; }
  sh[threadIdx.x] = s2;
  __syncthreads();
  for (int o = 128; o > 0; o >>= 1) {
    if (threadIdx.x < o) sh[threadIdx.x] += sh[threadIdx.x + o];
    __syncthreads();
  }
  const float rstd = rsqrtf(sh[0] / C + 1e-6f);
  cnt = 0;
  for (int c = threadIdx.x; c < C; c += 256) {
    const float v = (vloc[cnt++] - mean) * rstd * gamma[c] + beta[c];
    y[row * C + c] = v;
    if (ybf) ybf[row * C + c] = (__bf16)v;
  }
}

// GLU gate fuse: g = silu(t1) * t2 -> bf16
__global__ __launch_bounds__(256)
void k_glufuse(const float* __restrict__ t1, const float* __restrict__ t2,
               __bf16* __restrict__ g, long long n)
{
  const long long idx = (long long)blockIdx.x * 256 + threadIdx.x;
  if (idx >= n) return;
  const float a = t1[idx];
  const float sg = a / (1.f + __expf(-a));
  g[idx] = (__bf16)(sg * t2[idx]);
}

// Power-law graph weights: avAp = attw * sign(A)|A|^pw, written transposed bf16.
__global__ __launch_bounds__(256)
void k_powerlaw(const float* __restrict__ A, const float* __restrict__ pw,
                const float* __restrict__ attw, __bf16* __restrict__ outT)
{
  const int idx = blockIdx.x * 256 + threadIdx.x;  // 32*64*64
  const int e = idx & 63, d = (idx >> 6) & 63, bh = idx >> 12;
  const float a = A[idx];
  const float p = pw[d * 64 + e];
  const float w = attw[d * 64 + e];
  const float val = w * copysignf(__powf(fabsf(a), p), a);
  outT[(long long)bh * 4096 + e * 64 + d] = (__bf16)val;   // [bh][e][d]
}

// ---------------------------------------------------------------------------
// Host orchestration
// ---------------------------------------------------------------------------
extern "C" void kernel_launch(void* const* d_in, const int* in_sizes, int n_in,
                              void* d_out, int out_size, void* d_ws, size_t ws_size,
                              hipStream_t stream)
{
  (void)n_in; (void)out_size; (void)ws_size;

  const float* q    = (const float*)d_in[0];
  const float* k    = (const float*)d_in[1];
  const float* v    = (const float*)d_in[2];
  // d_in[3] = mask (unused: causal handled analytically)
  const float* wq_w = (const float*)d_in[4];  const float* wq_b = (const float*)d_in[5];
  const float* wk_w = (const float*)d_in[6];  const float* wk_b = (const float*)d_in[7];
  const float* wv_w = (const float*)d_in[8];  const float* wv_b = (const float*)d_in[9];
  const float* wo_w = (const float*)d_in[10]; const float* wo_b = (const float*)d_in[11];
  const float* ainp_w = (const float*)d_in[12]; const float* ainp_b = (const float*)d_in[13];
  const float* afin_w = (const float*)d_in[14]; const float* afin_b = (const float*)d_in[15];
  const float* ln1_g = (const float*)d_in[16]; const float* ln1_b = (const float*)d_in[17];
  const float* ln2_g = (const float*)d_in[18]; const float* ln2_b = (const float*)d_in[19];

  // res_params flatten: detect alphabetical-pytree vs insertion order via sizes
  const int resBase = 20;
  const bool alphaOrder = (in_sizes[resBase] < 100000);  // bias (1024) first => alpha
  const float* rw1[CNRES][CNGLU]; const float* rb1[CNRES][CNGLU];
  const float* rw2[CNRES][CNGLU]; const float* rb2[CNRES][CNGLU];
  const float* rw3[CNRES][CNGLU]; const float* rb3[CNRES][CNGLU];
  const float* rlg[CNRES]; const float* rlb[CNRES];
  for (int i = 0; i < CNRES; ++i) {
    const int bi = resBase + i * 14;
    for (int j = 0; j < CNGLU; ++j) {
      const int gj = bi + j * 6;
      if (alphaOrder) {
        rb1[i][j] = (const float*)d_in[gj + 0]; rb2[i][j] = (const float*)d_in[gj + 1];
        rb3[i][j] = (const float*)d_in[gj + 2]; rw1[i][j] = (const float*)d_in[gj + 3];
        rw2[i][j] = (const float*)d_in[gj + 4]; rw3[i][j] = (const float*)d_in[gj + 5];
      } else {
        rw1[i][j] = (const float*)d_in[gj + 0]; rb1[i][j] = (const float*)d_in[gj + 1];
        rw2[i][j] = (const float*)d_in[gj + 2]; rb2[i][j] = (const float*)d_in[gj + 3];
        rw3[i][j] = (const float*)d_in[gj + 4]; rb3[i][j] = (const float*)d_in[gj + 5];
      }
    }
    if (alphaOrder) { rlb[i] = (const float*)d_in[bi + 12]; rlg[i] = (const float*)d_in[bi + 13]; }
    else            { rlg[i] = (const float*)d_in[bi + 12]; rlb[i] = (const float*)d_in[bi + 13]; }
  }
  const int tailBase = resBase + CNRES * 14;
  const float* pw   = (const float*)d_in[tailBase + 0];
  const float* attw = (const float*)d_in[tailBase + 1];
  const float* ba   = (const float*)d_in[tailBase + 2];

  // ---- workspace carving -------------------------------------------------
  char* wp = (char*)d_ws;
  auto alloc = [&](size_t bytes) -> void* {
    void* r = (void*)wp;
    wp += (bytes + 255) & ~(size_t)255;
    return r;
  };
  const long long NBS = (long long)CB * CS;            // 4096 rows
  const long long NEL = NBS * CDM;                      // 4 Mi elements
  const long long NBH = (long long)CB * CH;             // 32
  const long long MET = NBH * CD;                       // 2048 metric rows

  __bf16* wqT  = (__bf16*)alloc(CDM * CDM * 2);
  __bf16* wkT  = (__bf16*)alloc(CDM * CDM * 2);
  __bf16* wvT  = (__bf16*)alloc(CDM * CDM * 2);
  __bf16* woT  = (__bf16*)alloc(CDM * CDM * 2);
  __bf16* ainpT = (__bf16*)alloc(CD * CAD * 2);
  __bf16* afinT = (__bf16*)alloc(CAD * CD * 2);
  __bf16* gw1T[CNRES][CNGLU]; __bf16* gw2T[CNRES][CNGLU]; __bf16* gw3T[CNRES][CNGLU];
  for (int i = 0; i < CNRES; ++i)
    for (int j = 0; j < CNGLU; ++j) {
      gw1T[i][j] = (__bf16*)alloc(CAD * CAD * 2);
      gw2T[i][j] = (__bf16*)alloc(CAD * CAD * 2);
      gw3T[i][j] = (__bf16*)alloc(CAD * CAD * 2);
    }
  __bf16* qbf = (__bf16*)alloc(NEL * 2);
  __bf16* kbf = (__bf16*)alloc(NEL * 2);
  __bf16* vbf = (__bf16*)alloc(NEL * 2);
  float*  qp  = (float*)alloc(NEL * 4);
  float*  kp  = (float*)alloc(NEL * 4);
  float*  vp  = (float*)alloc(NEL * 4);
  __bf16* qh   = (__bf16*)alloc(NEL * 2);
  __bf16* qhT  = (__bf16*)alloc(NEL * 2);
  __bf16* khb  = (__bf16*)alloc(NEL * 2);
  __bf16* vT   = (__bf16*)alloc(NEL * 2);
  __bf16* Ametric = (__bf16*)alloc(MET * CD * 2);
  float*  X1   = (float*)alloc(MET * CAD * 4);
  float*  bufA = (float*)alloc(MET * CAD * 4);
  float*  bufB = (float*)alloc(MET * CAD * 4);
  float*  bufC = (float*)alloc(MET * CAD * 4);
  __bf16* bufAbf = (__bf16*)alloc(MET * CAD * 2);
  __bf16* bufBbf = (__bf16*)alloc(MET * CAD * 2);
  __bf16* bufCbf = (__bf16*)alloc(MET * CAD * 2);
  float*  t1f  = (float*)alloc(MET * CAD * 4);
  float*  t2f  = (float*)alloc(MET * CAD * 4);
  __bf16* gbf  = (__bf16*)alloc(MET * CAD * 2);
  float*  Afin   = (float*)alloc(MET * CD * 4);
  float*  AfinLn = (float*)alloc(MET * CD * 4);
  __bf16* avApT  = (__bf16*)alloc(NBH * CD * CD * 2);
  __bf16* qA     = (__bf16*)alloc(NEL * 2);
  __bf16* Hout   = (__bf16*)alloc(NEL * 2);

  const dim3 T256(256);
  auto gemm = [&](const __bf16* A, const __bf16* BT, const float* bias,
                  float* C, __bf16* Cbf, int M, int N, int K,
                  long long sA, long long sBT, long long sC, int batch) {
    if ((M % 128 == 0) && (N % 128 == 0)) {
      dim3 grid(N / 128, M / 128, batch);
      k_gemm_bf16_128<<<grid, T256, 0, stream>>>(A, BT, bias, C, Cbf, M, N, K, sA, sBT, sC);
    } else {
      dim3 grid(N / 64, M / 64, batch);
      k_gemm_bf16<<<grid, T256, 0, stream>>>(A, BT, bias, C, Cbf, M, N, K, sA, sBT, sC);
    }
  };

  // ---- phase 0: weight transposes + input bf16 conversion ----------------
  const int gW = (CDM * CDM) / 256;
  k_convT<<<gW, T256, 0, stream>>>(wq_w, wqT, CDM, CDM);
  k_convT<<<gW, T256, 0, stream>>>(wk_w, wkT, CDM, CDM);
  k_convT<<<gW, T256, 0, stream>>>(wv_w, wvT, CDM, CDM);
  k_convT<<<gW, T256, 0, stream>>>(wo_w, woT, CDM, CDM);
  k_convT<<<(CD * CAD) / 256, T256, 0, stream>>>(ainp_w, ainpT, CD, CAD);
  k_convT<<<(CAD * CD) / 256, T256, 0, stream>>>(afin_w, afinT, CAD, CD);
  for (int i = 0; i < CNRES; ++i)
    for (int j = 0; j < CNGLU; ++j) {
      k_convT<<<gW, T256, 0, stream>>>(rw1[i][j], gw1T[i][j], CAD, CAD);
      k_convT<<<gW, T256, 0, stream>>>(rw2[i][j], gw2T[i][j], CAD, CAD);
      k_convT<<<gW, T256, 0, stream>>>(rw3[i][j], gw3T[i][j], CAD, CAD);
    }
  const int gE = (int)(NEL / 256);
  k_conv<<<gE, T256, 0, stream>>>(q, qbf, NEL);
  k_conv<<<gE, T256, 0, stream>>>(k, kbf, NEL);
  k_conv<<<gE, T256, 0, stream>>>(v, vbf, NEL);

  // ---- phase 1: QKV projections ------------------------------------------
  gemm(qbf, wqT, wq_b, qp, nullptr, (int)NBS, CDM, CDM, 0, 0, 0, 1);
  gemm(kbf, wkT, wk_b, kp, nullptr, (int)NBS, CDM, CDM, 0, 0, 0, 1);
  gemm(vbf, wvT, wv_b, vp, nullptr, (int)NBS, CDM, CDM, 0, 0, 0, 1);

  // ---- phase 2: RoPE + head split / transposes ---------------------------
  k_qkv_prep<<<gE, T256, 0, stream>>>(qp, kp, vp, qh, qhT, khb, vT);

  // ---- phase 3: metric tensor A = qh^T qh  (batched 64x64, K=2048) -------
  gemm(qhT, qhT, nullptr, nullptr, Ametric, CD, CD, CS,
       (long long)CD * CS, (long long)CD * CS, (long long)CD * CD, (int)NBH);

  // ---- phase 4: A @ ainp_w + b, LN1 --------------------------------------
  gemm(Ametric, ainpT, ainp_b, X1, nullptr, (int)MET, CAD, CD, 0, 0, 0, 1);
  k_ln<<<(int)MET, T256, 0, stream>>>(X1, nullptr, ln1_g, ln1_b, bufA, bufAbf, CAD);

  // ---- phase 5: residual GLU chain ---------------------------------------
  const long long NG = MET * CAD;
  const int gG = (int)(NG / 256);
  for (int i = 0; i < CNRES; ++i) {
    const __bf16* inbf = bufAbf;
    for (int j = 0; j < CNGLU; ++j) {
      float*  outF  = (j == 0) ? bufB  : bufC;
      __bf16* outBF = (j == 0) ? bufBbf : bufCbf;
      gemm(inbf, gw1T[i][j], rb1[i][j], t1f, nullptr, (int)MET, CAD, CAD, 0, 0, 0, 1);
      gemm(inbf, gw2T[i][j], rb2[i][j], t2f, nullptr, (int)MET, CAD, CAD, 0, 0, 0, 1);
      k_glufuse<<<gG, T256, 0, stream>>>(t1f, t2f, gbf, NG);
      gemm(gbf, gw3T[i][j], rb3[i][j], outF, outBF, (int)MET, CAD, CAD, 0, 0, 0, 1);
      inbf = outBF;
    }
    // A = LN(glu_out + A_in)  (residual in bufA), back into bufA
    k_ln<<<(int)MET, T256, 0, stream>>>(bufC, bufA, rlg[i], rlb[i], bufA, bufAbf, CAD);
  }

  // ---- phase 6: A @ afin_w + b, LN2 --------------------------------------
  gemm(bufAbf, afinT, afin_b, Afin, nullptr, (int)MET, CD, CAD, 0, 0, 0, 1);
  k_ln<<<(int)MET, T256, 0, stream>>>(Afin, nullptr, ln2_g, ln2_b, AfinLn, nullptr, CD);

  // ---- phase 7: power-law graph weights ----------------------------------
  k_powerlaw<<<(int)(NBH * CD * CD / 256), T256, 0, stream>>>(AfinLn, pw, attw, avApT);

  // ---- phase 8: qA = qh @ avAp (batched) ---------------------------------
  gemm(qh, avApT, nullptr, nullptr, qA, CS, CD, CD,
       (long long)CS * CD, (long long)CD * CD, (long long)CS * CD, (int)NBH);

  // ---- phase 9: causal flash attention -----------------------------------
  {
    dim3 grid(CS / 64, (unsigned)NBH);
    k_flash<<<grid, T256, 0, stream>>>(qA, khb, vT, ba, Hout);
  }

  // ---- phase 10: output projection -> d_out (f32) ------------------------
  gemm(Hout, woT, wo_b, (float*)d_out, nullptr, (int)NBS, CDM, CDM, 0, 0, 0, 1);
}
